// FusedLlamaLowBitDecoderlayer_73796128080359
// MI455X (gfx1250) — compile-verified, matrix operand order checked by reading, not run on hardware
//
#include <hip/hip_runtime.h>
#include <cstdint>

// ---------------------------------------------------------------------------
// MI455X (gfx1250) fused Llama decoder layer.
// Compute-bound (~0.9 TFLOP vs ~45us of HBM traffic) => all GEMMs + attention
// run on v_wmma_f32_16x16x32_bf16. fp32 weights are converted to bf16 with
// native v_cvt_pk_bf16_f32 during global->LDS staging (read-once => no extra
// HBM traffic). bf16-clean tiles (GEMM A, attention K) use
// GLOBAL_LOAD_ASYNC_TO_LDS_B128 (ASYNCcnt) to bypass the VGPR round-trip;
// drains sit after the WMMA block so copy latency hides under matrix math.
// ---------------------------------------------------------------------------

typedef __attribute__((ext_vector_type(16))) __bf16 v16bf;
typedef __attribute__((ext_vector_type(8)))  float  v8f;

// Native RNE f32->bf16 (hardware cvt on gfx1250).
__device__ __forceinline__ unsigned short f2bf(float f) {
    __bf16 h = (__bf16)f;
    unsigned short u;
    __builtin_memcpy(&u, &h, 2);
    return u;
}

// Pack two f32 into one dword of (lo,hi) bf16 with a single packed convert.
// (Mnemonic verified in gfx1250 codegen; inline asm pins the operand pairing
// that the SLP vectorizer otherwise scrambles.)
__device__ __forceinline__ unsigned pk2bf(float lo, float hi) {
    unsigned r;
    asm("v_cvt_pk_bf16_f32 %0, %1, %2" : "=v"(r) : "v"(lo), "v"(hi));
    return r;
}

// Async global->LDS 16-byte copy (per lane). Generic shared pointers carry the
// LDS byte offset in their low 32 bits (flat LDS aperture).
__device__ __forceinline__ void async_b128(const void* g, void* l) {
    const unsigned loff = (unsigned)(uintptr_t)l;
    asm volatile("global_load_async_to_lds_b128 %0, %1, off"
                 :: "v"(loff), "v"(g) : "memory");
}
__device__ __forceinline__ void wait_async0() {
    asm volatile("s_wait_asynccnt 0x0" ::: "memory");
}

// Build a bf16 WMMA fragment for wave32.
// Per ISA 7.12.2: lanes<16 hold K = kbase..kbase+7 and kbase+16..kbase+23,
// lanes>=16 the +8 offsets; caller folds the lane-half into `kbase`.
__device__ __forceinline__ v16bf ld_frag(const unsigned short* p, int kbase) {
    union { v16bf v; uint4 q[2]; } u;
    u.q[0] = *reinterpret_cast<const uint4*>(p + kbase);
    u.q[1] = *reinterpret_cast<const uint4*>(p + kbase + 16);
    return u.v;
}

// ---------------------------------------------------------------------------
// RMSNorm over H=4096, writes bf16 normalized row.
// ---------------------------------------------------------------------------
__global__ void __launch_bounds__(256) rmsnorm_cast(
    const float* __restrict__ x, const float* __restrict__ w,
    unsigned short* __restrict__ out, int H)
{
    __shared__ float red[256];
    const int s = blockIdx.x, t = threadIdx.x;
    const float4* xp = reinterpret_cast<const float4*>(x + (size_t)s * H);
    float4 v[4];
    float ss = 0.f;
#pragma unroll
    for (int i = 0; i < 4; ++i) {
        v[i] = xp[t + 256 * i];
        ss += v[i].x * v[i].x + v[i].y * v[i].y + v[i].z * v[i].z + v[i].w * v[i].w;
    }
    red[t] = ss;
    __syncthreads();
    for (int st = 128; st > 0; st >>= 1) {
        if (t < st) red[t] += red[t + st];
        __syncthreads();
    }
    const float scale = rsqrtf(red[0] / (float)H + 1e-5f);
    const float4* wp = reinterpret_cast<const float4*>(w);
    unsigned* op = reinterpret_cast<unsigned*>(out + (size_t)s * H);
#pragma unroll
    for (int i = 0; i < 4; ++i) {
        float4 ww = wp[t + 256 * i];
        const int base = (t + 256 * i) * 2;       // in 32-bit units
        op[base + 0] = pk2bf(v[i].x * scale * ww.x, v[i].y * scale * ww.y);
        op[base + 1] = pk2bf(v[i].z * scale * ww.z, v[i].w * scale * ww.w);
    }
}

// ---------------------------------------------------------------------------
// C[M,N](f32) = A[M,K](bf16) @ W[K,N](f32->bf16 in LDS) [+ Res]
// 128x128x32 tile, 256 threads = 8 waves, wave = 32x64 C region.
// Pipeline per k-step: issue async-A(next) + load-B-regs(next) -> WMMA(cur)
// -> convert/pack/store B(next) -> s_wait_asynccnt -> barrier.
// B is staged transposed [n][k]; each thread owns a k-pair x 8 columns so the
// transposed stores are packed ds_store_b32 (cvt_pk pairs k with k+1).
// ---------------------------------------------------------------------------
#define BM 128
#define BN 128
#define BK 32
#define BKP 40   // padded LDS row stride (shorts), 80B = multiple of 16B

__global__ void __launch_bounds__(256) gemm_bf16f32(
    const unsigned short* __restrict__ A, const float* __restrict__ W,
    float* __restrict__ C, const float* __restrict__ Res,
    int M, int N, int K)
{
    __shared__ __align__(16) unsigned short As[2][BM][BKP];
    __shared__ __align__(16) unsigned short Bs[2][BN][BKP];

    const int t = threadIdx.x;
    const int lane = t & 31, wave = t >> 5;
    const int wr = wave >> 1, wc = wave & 1;          // 4x2 wave grid
    const int lm = lane & 15, hi = lane >> 4;
    const int kb = hi << 3;
    const int row0 = blockIdx.y * BM;
    const int col0 = blockIdx.x * BN;

    v8f acc[2][4];
#pragma unroll
    for (int mt = 0; mt < 2; ++mt)
#pragma unroll
        for (int nt = 0; nt < 4; ++nt) acc[mt][nt] = (v8f){};

    const int ar   = t >> 1, ah = (t & 1) * 16;       // A: 2 thr/row, 16 bf16 each
    const int bkp2 = (t & 15) * 2;                    // B: k-pair base
    const int bc8  = (t >> 4) * 8;                    // B: 8 columns

    const unsigned short* Arow = A + (size_t)(row0 + ar) * K + ah;
    const float*          Wrow = W + (size_t)bkp2 * N + col0 + bc8;

    auto issue_A = [&](int k0, int buf) {
        const unsigned short* g = Arow + k0;
        async_b128(g,     &As[buf][ar][ah]);
        async_b128(g + 8, &As[buf][ar][ah + 8]);
    };
    auto load_B = [&](int k0, float4* f) {
        const float* w0 = Wrow + (size_t)k0 * N;       // row k
        const float* w1 = w0 + N;                      // row k+1
        f[0] = reinterpret_cast<const float4*>(w0)[0];
        f[1] = reinterpret_cast<const float4*>(w0)[1];
        f[2] = reinterpret_cast<const float4*>(w1)[0];
        f[3] = reinterpret_cast<const float4*>(w1)[1];
    };
    auto store_B = [&](int buf, const float4* f) {
        const float a[8] = { f[0].x, f[0].y, f[0].z, f[0].w,
                             f[1].x, f[1].y, f[1].z, f[1].w };
        const float b[8] = { f[2].x, f[2].y, f[2].z, f[2].w,
                             f[3].x, f[3].y, f[3].z, f[3].w };
#pragma unroll
        for (int i = 0; i < 8; ++i)
            *reinterpret_cast<unsigned*>(&Bs[buf][bc8 + i][bkp2]) =
                pk2bf(a[i], b[i]);
    };

    float4 fB[4];
    issue_A(0, 0);
    load_B(0, fB);
    store_B(0, fB);
    wait_async0();
    __syncthreads();

    const int nk = K / BK;
    for (int kk = 0; kk < nk; ++kk) {
        const int buf = kk & 1;
        const bool more = (kk + 1 < nk);
        if (more) {                         // stage next tile
            issue_A((kk + 1) * BK, buf ^ 1);
            load_B((kk + 1) * BK, fB);
        }

        v16bf af[2], bfv[4];                // compute current tile
#pragma unroll
        for (int mt = 0; mt < 2; ++mt)
            af[mt] = ld_frag(&As[buf][wr * 32 + mt * 16 + lm][0], kb);
#pragma unroll
        for (int nt = 0; nt < 4; ++nt)
            bfv[nt] = ld_frag(&Bs[buf][wc * 64 + nt * 16 + lm][0], kb);
#pragma unroll
        for (int mt = 0; mt < 2; ++mt)
#pragma unroll
            for (int nt = 0; nt < 4; ++nt)
                acc[mt][nt] = __builtin_amdgcn_wmma_f32_16x16x32_bf16(
                    false, af[mt], false, bfv[nt], (short)0, acc[mt][nt],
                    false, false);

        if (more) store_B(buf ^ 1, fB);
        wait_async0();
        __syncthreads();
    }

    // Epilogue: C-layout (row = r + 8*hi, col = lane%16)
#pragma unroll
    for (int mt = 0; mt < 2; ++mt) {
#pragma unroll
        for (int nt = 0; nt < 4; ++nt) {
            const int c = col0 + wc * 64 + nt * 16 + lm;
            const int rbase = row0 + wr * 32 + mt * 16 + hi * 8;
#pragma unroll
            for (int r = 0; r < 8; ++r) {
                const size_t idx = (size_t)(rbase + r) * N + c;
                float v = acc[mt][nt][r];
                if (Res) v += Res[idx];
                C[idx] = v;
            }
        }
    }
}

// ---------------------------------------------------------------------------
// RoPE + head-major scatter; also emits fp32 k,v into d_out and bf16 q,k,v
// for the attention kernel.  grid(S, NH), 128 threads = one head-row.
// ---------------------------------------------------------------------------
__global__ void __launch_bounds__(128) rope_scatter(
    const float* __restrict__ qf, const float* __restrict__ kf,
    const float* __restrict__ vf, const float* __restrict__ cosT,
    const float* __restrict__ sinT, const int* __restrict__ pos,
    unsigned short* __restrict__ qb, unsigned short* __restrict__ kb16,
    unsigned short* __restrict__ vb, float* __restrict__ kout,
    float* __restrict__ vout, int S)
{
    const int s = blockIdx.x, h = blockIdx.y, j = threadIdx.x;
    const int p = pos[s];
    const float c  = cosT[p * 128 + j];
    const float sn = sinT[p * 128 + j];
    const size_t src = (size_t)s * 4096 + h * 128;
    const float q0 = qf[src + j];
    const float k0 = kf[src + j];
    float qr, kr;
    if (j < 64) { qr = q0 * c - qf[src + j + 64] * sn; kr = k0 * c - kf[src + j + 64] * sn; }
    else        { qr = q0 * c + qf[src + j - 64] * sn; kr = k0 * c + kf[src + j - 64] * sn; }
    const size_t dst = ((size_t)h * S + s) * 128 + j;
    qb[dst]   = f2bf(qr);
    kb16[dst] = f2bf(kr);
    kout[dst] = kr;
    const float v0 = vf[src + j];
    vb[dst]   = f2bf(v0);
    vout[dst] = v0;
}

// ---------------------------------------------------------------------------
// Flash attention (causal), one block = (head, 64 q rows), 4 waves, each wave
// owns 16 q rows.  QK^T and PV via bf16 WMMA; online softmax with 16-lane
// shuffle row-reductions; P transposed C->A layout through wave-private LDS.
// K tile staged via async-to-LDS; V transposed with key-pair packed b32 stores.
// ---------------------------------------------------------------------------
__global__ void __launch_bounds__(128) flash_attn(
    const unsigned short* __restrict__ Qb, const unsigned short* __restrict__ Kb,
    const unsigned short* __restrict__ Vb, unsigned short* __restrict__ Ob, int S)
{
    __shared__ __align__(16) unsigned short Kt[64][136];  // [key][d]
    __shared__ __align__(16) unsigned short Vt[128][72];  // [d][key] (transposed)
    __shared__ __align__(16) unsigned short Ps[4][16][72];

    const int t = threadIdx.x, lane = t & 31, wave = t >> 5;
    const int lm = lane & 15, hi = lane >> 4;
    const int kb = hi << 3, hi8 = hi << 3;
    const int qt = blockIdx.x, head = blockIdx.y;
    const size_t hbase = (size_t)head * S * 128;

    // Preload Q A-fragments (16 rows x 128 K, 4 chunks of K=32)
    v16bf qfr[4];
    {
        const unsigned short* qp =
            Qb + hbase + (size_t)(qt * 64 + wave * 16 + lm) * 128;
#pragma unroll
        for (int c = 0; c < 4; ++c) qfr[c] = ld_frag(qp, c * 32 + kb);
    }

    v8f O[8];
#pragma unroll
    for (int nt = 0; nt < 8; ++nt) O[nt] = (v8f){};
    float mrow[8], lrow[8];
#pragma unroll
    for (int r = 0; r < 8; ++r) { mrow[r] = -3.0e38f; lrow[r] = 0.f; }

    const int krw = t & 63, seg = t >> 6;    // K loader mapping
    const int kp  = (t & 31) * 2;            // V loader: key pair
    const int ds0 = (t >> 5) * 32;           // V loader: 32 d values

    for (int j = 0; j <= qt; ++j) {
        __syncthreads();
        {   // K tile: async bf16 copy (64x128), 4x16B per thread
            const unsigned short* kg =
                Kb + hbase + (size_t)(j * 64 + krw) * 128 + seg * 64;
            unsigned short* kl = &Kt[krw][seg * 64];
#pragma unroll
            for (int i = 0; i < 4; ++i) async_b128(kg + i * 8, kl + i * 8);
            // V tile: transpose key-pairs into Vt[d][key] with packed b32 stores
            const uint4* v0 = reinterpret_cast<const uint4*>(
                Vb + hbase + (size_t)(j * 64 + kp) * 128 + ds0);
            const uint4* v1 = reinterpret_cast<const uint4*>(
                Vb + hbase + (size_t)(j * 64 + kp + 1) * 128 + ds0);
#pragma unroll
            for (int i = 0; i < 4; ++i) {               // 4 x uint4 = 32 shorts
                const uint4 a = v0[i];
                const uint4 b = v1[i];
                const unsigned aw[4] = { a.x, a.y, a.z, a.w };
                const unsigned bw[4] = { b.x, b.y, b.z, b.w };
#pragma unroll
                for (int w2 = 0; w2 < 4; ++w2) {        // each word = 2 d's
                    const int d = ds0 + i * 8 + w2 * 2;
                    const unsigned lo = (aw[w2] & 0xFFFFu) | (bw[w2] << 16);
                    const unsigned hh = (aw[w2] >> 16) | (bw[w2] & 0xFFFF0000u);
                    *reinterpret_cast<unsigned*>(&Vt[d][kp])     = lo;
                    *reinterpret_cast<unsigned*>(&Vt[d + 1][kp]) = hh;
                }
            }
        }
        wait_async0();
        __syncthreads();

        // S = Q K^T (16x64 per wave)
        float sc[4][8];
#pragma unroll
        for (int nt = 0; nt < 4; ++nt) {
            v8f s = (v8f){};
#pragma unroll
            for (int c = 0; c < 4; ++c) {
                v16bf kfrag = ld_frag(&Kt[nt * 16 + lm][0], c * 32 + kb);
                s = __builtin_amdgcn_wmma_f32_16x16x32_bf16(
                    false, qfr[c], false, kfrag, (short)0, s, false, false);
            }
            const int col = j * 64 + nt * 16 + lm;
#pragma unroll
            for (int r = 0; r < 8; ++r) {
                float v = s[r] * 0.08838834764831845f;   // 1/sqrt(128)
                const int row = qt * 64 + wave * 16 + hi8 + r;
                sc[nt][r] = (col > row) ? -3.0e38f : v;
            }
        }

        // Online softmax (rows live in one 16-lane half)
#pragma unroll
        for (int r = 0; r < 8; ++r) {
            float rm = fmaxf(fmaxf(sc[0][r], sc[1][r]), fmaxf(sc[2][r], sc[3][r]));
#pragma unroll
            for (int off = 1; off < 16; off <<= 1)
                rm = fmaxf(rm, __shfl_xor(rm, off, 32));
            const float nm = fmaxf(mrow[r], rm);
            const float corr = __expf(mrow[r] - nm);
            float rs = 0.f;
#pragma unroll
            for (int nt = 0; nt < 4; ++nt) {
                float p = __expf(sc[nt][r] - nm);
                sc[nt][r] = p; rs += p;
            }
#pragma unroll
            for (int off = 1; off < 16; off <<= 1)
                rs += __shfl_xor(rs, off, 32);
            lrow[r] = lrow[r] * corr + rs;
            mrow[r] = nm;
#pragma unroll
            for (int nt = 0; nt < 8; ++nt) O[nt][r] *= corr;
        }

        // P: C-layout -> A-layout through wave-private LDS (in-order per wave)
#pragma unroll
        for (int nt = 0; nt < 4; ++nt)
#pragma unroll
            for (int r = 0; r < 8; ++r)
                Ps[wave][hi8 + r][nt * 16 + lm] = f2bf(sc[nt][r]);

        // O += P V   (K-dim = 64 keys = 2 chunks)
#pragma unroll
        for (int c = 0; c < 2; ++c) {
            v16bf pf = ld_frag(&Ps[wave][lm][0], c * 32 + kb);
#pragma unroll
            for (int nt = 0; nt < 8; ++nt) {
                v16bf vfrag = ld_frag(&Vt[nt * 16 + lm][0], c * 32 + kb);
                O[nt] = __builtin_amdgcn_wmma_f32_16x16x32_bf16(
                    false, pf, false, vfrag, (short)0, O[nt], false, false);
            }
        }
    }

    // Normalize and store bf16 row-major [S][4096] for the WO GEMM
#pragma unroll
    for (int nt = 0; nt < 8; ++nt) {
        const int c = nt * 16 + lm;
#pragma unroll
        for (int r = 0; r < 8; ++r) {
            const int row = qt * 64 + wave * 16 + hi8 + r;
            const float v = O[nt][r] / lrow[r];
            Ob[(size_t)row * 4096 + head * 128 + c] = f2bf(v);
        }
    }
}

// ---------------------------------------------------------------------------
// SwiGLU elementwise: bf16(silu(gate) * up), packed b32 stores.
// ---------------------------------------------------------------------------
__global__ void __launch_bounds__(256) silu_mul(
    const float* __restrict__ g, const float* __restrict__ u,
    unsigned short* __restrict__ out, int n)
{
    const int i = (blockIdx.x * 256 + threadIdx.x) * 4;
    if (i >= n) return;
    const float4 gv = *reinterpret_cast<const float4*>(g + i);
    const float4 uv = *reinterpret_cast<const float4*>(u + i);
    const float a0 = gv.x / (1.f + __expf(-gv.x)) * uv.x;
    const float a1 = gv.y / (1.f + __expf(-gv.y)) * uv.y;
    const float a2 = gv.z / (1.f + __expf(-gv.z)) * uv.z;
    const float a3 = gv.w / (1.f + __expf(-gv.w)) * uv.w;
    unsigned* op = reinterpret_cast<unsigned*>(out + i);
    op[0] = pk2bf(a0, a1);
    op[1] = pk2bf(a2, a3);
}

// ---------------------------------------------------------------------------
extern "C" void kernel_launch(void* const* d_in, const int* in_sizes, int n_in,
                              void* d_out, int out_size, void* d_ws, size_t ws_size,
                              hipStream_t stream)
{
    const float* hs    = (const float*)d_in[0];
    const int*   pos   = (const int*)  d_in[1];
    const float* ln0w  = (const float*)d_in[2];
    const float* ln1w  = (const float*)d_in[3];
    const float* wq    = (const float*)d_in[4];
    const float* wk    = (const float*)d_in[5];
    const float* wv    = (const float*)d_in[6];
    const float* wo    = (const float*)d_in[7];
    const float* wgate = (const float*)d_in[8];
    const float* wup   = (const float*)d_in[9];
    const float* wdown = (const float*)d_in[10];
    const float* cosT  = (const float*)d_in[11];
    const float* sinT  = (const float*)d_in[12];

    const int S = 2048, H = 4096, I = 11008, NH = 32;

    char* ws = (char*)d_ws;
    size_t off = 0;
    auto alloc = [&](size_t bytes) -> void* {
        void* p = ws + off;
        off = (off + bytes + 255) & ~(size_t)255;
        return p;
    };
    unsigned short* xn      = (unsigned short*)alloc((size_t)S * H * 2);
    float*          qf      = (float*)         alloc((size_t)S * H * 4);
    float*          kf      = (float*)         alloc((size_t)S * H * 4);
    float*          vf      = (float*)         alloc((size_t)S * H * 4);
    unsigned short* qb      = (unsigned short*)alloc((size_t)S * H * 2);
    unsigned short* kb16    = (unsigned short*)alloc((size_t)S * H * 2);
    unsigned short* vb      = (unsigned short*)alloc((size_t)S * H * 2);
    unsigned short* attnb   = (unsigned short*)alloc((size_t)S * H * 2);
    float*          hidden1 = (float*)         alloc((size_t)S * H * 4);
    unsigned short* yb      = (unsigned short*)alloc((size_t)S * H * 2);
    float*          gatef   = (float*)         alloc((size_t)S * I * 4);
    float*          upf     = (float*)         alloc((size_t)S * I * 4);
    unsigned short* actb    = (unsigned short*)alloc((size_t)S * I * 2);

    float* out_h = (float*)d_out;                 // [S,H]
    float* out_k = out_h + (size_t)S * H;         // [NH,S,128]
    float* out_v = out_k + (size_t)S * H;         // [NH,S,128]

    const dim3 gH(H / BN, S / BM);   // N=4096 GEMMs
    const dim3 gI(I / BN, S / BM);   // N=11008 GEMMs

    // 1) pre-attn RMSNorm
    rmsnorm_cast<<<S, 256, 0, stream>>>(hs, ln0w, xn, H);
    // 2) QKV projections
    gemm_bf16f32<<<gH, 256, 0, stream>>>(xn, wq, qf, nullptr, S, H, H);
    gemm_bf16f32<<<gH, 256, 0, stream>>>(xn, wk, kf, nullptr, S, H, H);
    gemm_bf16f32<<<gH, 256, 0, stream>>>(xn, wv, vf, nullptr, S, H, H);
    // 3) RoPE + head-major scatter (+ fp32 k/v outputs)
    rope_scatter<<<dim3(S, NH), 128, 0, stream>>>(qf, kf, vf, cosT, sinT, pos,
                                                  qb, kb16, vb, out_k, out_v, S);
    // 4) causal flash attention
    flash_attn<<<dim3(S / 64, NH), 128, 0, stream>>>(qb, kb16, vb, attnb, S);
    // 5) output projection + residual
    gemm_bf16f32<<<gH, 256, 0, stream>>>(attnb, wo, hidden1, hs, S, H, H);
    // 6) post-attn RMSNorm
    rmsnorm_cast<<<S, 256, 0, stream>>>(hidden1, ln1w, yb, H);
    // 7) SwiGLU MLP
    gemm_bf16f32<<<gI, 256, 0, stream>>>(yb, wgate, gatef, nullptr, S, I, H);
    gemm_bf16f32<<<gI, 256, 0, stream>>>(yb, wup,   upf,   nullptr, S, I, H);
    silu_mul<<<((S * I / 4) + 255) / 256, 256, 0, stream>>>(gatef, upf, actb, S * I);
    gemm_bf16f32<<<gH, 256, 0, stream>>>(actb, wdown, out_h, hidden1, S, H, I);
}
